// DRMM_76175539961937
// MI455X (gfx1250) — matrix operand order repaired and checked by reading, not compiled, not run on hardware
//
#include <hip/hip_runtime.h>
#include <hip/hip_bf16.h>

#define VOCAB     50000
#define EMBED_DIM 300
#define EPAD      320      // K padded to multiple of 32 for WMMA
#define NBINS     30
#define NBINS_PAD 32
#define BATCH     32
#define QLEN      16
#define DLEN      2048
#define DTILE     64       // docs per block (4 waves x 16 cols)

typedef __attribute__((ext_vector_type(16))) _Float16 v16h;
typedef __attribute__((ext_vector_type(8)))  _Float16 v8h;
typedef __attribute__((ext_vector_type(8)))  float    v8f;

// ---------------------------------------------------------------------------
// Kernel A: per-(b,q) row norm + gate logit; write qn as f16 (K padded to 320);
// zero the global histogram. One wave (32 lanes) per query token.
// ---------------------------------------------------------------------------
__global__ void __launch_bounds__(512)
drmm_prep(const int* __restrict__ query,
          const float* __restrict__ emb,
          const float* __restrict__ Wg,
          const float* __restrict__ bg,
          _Float16* __restrict__ qn,
          float* __restrict__ gate,
          float* __restrict__ qmk,
          unsigned* __restrict__ hist) {
  const int b   = blockIdx.x;
  const int tid = threadIdx.x;

  // zero hist: 16*32 = 512 entries per batch, 512 threads
  hist[(size_t)b * QLEN * NBINS_PAD + tid] = 0u;

  const int q    = tid >> 5;
  const int lane = tid & 31;
  const int id   = query[b * QLEN + q];
  const float mask = (id > 0) ? 1.0f : 0.0f;
  const float* row = emb + (size_t)id * EMBED_DIM;

  float ss = 0.0f, dg = 0.0f;
  for (int e = lane; e < EMBED_DIM; e += 32) {
    float v = row[e] * mask;
    ss += v * v;
    dg += v * Wg[e];
  }
  #pragma unroll
  for (int m = 16; m >= 1; m >>= 1) {
    ss += __shfl_xor(ss, m, 32);
    dg += __shfl_xor(dg, m, 32);
  }
  const float rn = rsqrtf(ss);   // pad row -> inf -> NaN sims, dropped by hist

  if (lane == 0) {
    gate[b * QLEN + q] = tanhf(dg + bg[0]);
    qmk [b * QLEN + q] = mask;
  }

  _Float16* out = qn + (size_t)(b * QLEN + q) * EPAD;
  for (int e = lane; e < EPAD; e += 32) {
    float v = (e < EMBED_DIM) ? row[e] * mask * rn : 0.0f;
    out[e] = (_Float16)v;
  }
}

// ---------------------------------------------------------------------------
// Kernel B: cosine interaction via WMMA f16 + histogram.
// Block = (batch b, 64-doc tile). 4 waves; each wave owns a 16-doc column tile
// and computes a 16(q) x 16(d) f32 accumulator over K=320 in 10 WMMA steps.
// ---------------------------------------------------------------------------
__global__ void __launch_bounds__(128)
drmm_interact(const int* __restrict__ document,
              const float* __restrict__ emb,
              const _Float16* __restrict__ qn,
              unsigned* __restrict__ hist) {
  __shared__ _Float16 qs [QLEN ][EPAD];     // 10 KB
  __shared__ _Float16 dsm[DTILE][EPAD];     // 40 KB
  __shared__ unsigned lhist[QLEN][NBINS_PAD]; // 2 KB

  const int b     = blockIdx.y;
  const int dbase = blockIdx.x * DTILE;
  const int tid   = threadIdx.x;
  const int lane  = tid & 31;
  const int wave  = tid >> 5;

  for (int i = tid; i < QLEN * NBINS_PAD; i += 128) ((unsigned*)lhist)[i] = 0u;

  // Stage normalized query tile (already f16 in workspace): 16x320 halfs
  {
    const uint4* src = (const uint4*)(qn + (size_t)b * QLEN * EPAD);
    uint4*       dst = (uint4*)&qs[0][0];
    for (int i = tid; i < QLEN * EPAD / 8; i += 128) dst[i] = src[i];
  }

  // Stage 64 document embedding rows -> f16 LDS (zero-padded K 300..319).
  // 2 threads per row; float4 gathers (emb rows are 1200B -> 16B aligned).
  {
    const int j    = tid >> 1;
    const int half = tid & 1;
    const int id   = document[(size_t)b * DLEN + dbase + j];
    const float mask = (id > 0) ? 1.0f : 0.0f;
    const float* row = emb + (size_t)id * EMBED_DIM;
    __builtin_prefetch(row + half * 160, 0, 0);
    #pragma unroll 4
    for (int v = 0; v < 40; ++v) {
      const int e = half * 160 + v * 4;            // 300 % 4 == 0: all-or-none
      float4 f = make_float4(0.f, 0.f, 0.f, 0.f);
      if (e < EMBED_DIM) f = *(const float4*)(row + e);
      dsm[j][e + 0] = (_Float16)(f.x * mask);
      dsm[j][e + 1] = (_Float16)(f.y * mask);
      dsm[j][e + 2] = (_Float16)(f.z * mask);
      dsm[j][e + 3] = (_Float16)(f.w * mask);
    }
  }
  __syncthreads();

  // Column (doc) for this lane per the 16x16 C layout: N = lane % 16.
  const int n   = lane & 15;
  const int doc = wave * 16 + n;

  // 1/||d|| for this lane's column (pair of lanes split the K range).
  float rn;
  {
    const int half = lane >> 4;
    float ss = 0.0f;
    for (int e = half * 160; e < half * 160 + 160; ++e) {
      float v = (float)dsm[doc][e];
      ss += v * v;
    }
    ss += __shfl_xor(ss, 16, 32);
    rn = rsqrtf(ss);
  }

  // WMMA fragment addressing (ISA 7.12.2, wave32, 16-bit 16x32 A / 32x16 B):
  //  A: lanes 0-15 row M=lane, K {0..7,16..23}; lanes 16-31 same rows, K {8..15,24..31}
  //  B: lane -> column N=lane%16; lanes 0-15 K {0..7,16..23}, lanes 16-31 K {8..15,24..31}
  const int m    = lane & 15;
  const int aklo = (lane < 16) ? 0 : 8;
  const int bklo = (lane < 16) ? 0 : 8;

  v8f acc = {};
  #pragma unroll
  for (int ks = 0; ks < EPAD / 32; ++ks) {
    const int kb = ks * 32;
    v8h a0 = *(const v8h*)&qs [m  ][kb + aklo];
    v8h a1 = *(const v8h*)&qs [m  ][kb + aklo + 16];
    v8h b0 = *(const v8h*)&dsm[doc][kb + bklo];
    v8h b1 = *(const v8h*)&dsm[doc][kb + bklo + 16];
    v16h a, bb;
    #pragma unroll
    for (int i = 0; i < 8; ++i) {
      a [i] = a0[i]; a [i + 8] = a1[i];
      bb[i] = b0[i]; bb[i + 8] = b1[i];
    }
    acc = __builtin_amdgcn_wmma_f32_16x16x32_f16(
        /*neg_a=*/false, a, /*neg_b=*/false, bb,
        /*c_mod=*/(short)0, acc, /*reuse_a=*/false, /*reuse_b=*/false);
  }

  // Scale by 1/||d|| (A rows already normalized) and bin into [-1,1] hist.
  const int qhi = (lane >= 16) ? 8 : 0;   // VGPR r -> query row r (+8 hi lanes)
  #pragma unroll
  for (int r = 0; r < 8; ++r) {
    float x = acc[r] * rn;
    if (x >= -1.0f && x <= 1.0f) {        // NaN (pad rows) fails -> dropped
      int bin = (int)floorf((x + 1.0f) * (NBINS * 0.5f));
      bin = bin < 0 ? 0 : (bin > NBINS - 1 ? NBINS - 1 : bin);
      atomicAdd(&lhist[r + qhi][bin], 1u);
    }
  }
  __syncthreads();

  unsigned* gh = hist + (size_t)b * QLEN * NBINS_PAD;
  for (int i = tid; i < QLEN * NBINS_PAD; i += 128) {
    unsigned v = ((unsigned*)lhist)[i];
    if (v) atomicAdd(&gh[i], v);
  }
}

// ---------------------------------------------------------------------------
// Kernel C: log1p(hist) -> tiny tanh MLP -> masked softmax gate -> sum over Q.
// One wave per batch; lanes 0..15 each own one query token.
// ---------------------------------------------------------------------------
__global__ void __launch_bounds__(32)
drmm_head(const unsigned* __restrict__ hist,
          const float* __restrict__ gate,
          const float* __restrict__ qmk,
          const float* __restrict__ W1, const float* __restrict__ b1,
          const float* __restrict__ W2, const float* __restrict__ b2,
          const float* __restrict__ W3, const float* __restrict__ b3,
          float* __restrict__ out) {
  const int b = blockIdx.x;
  const int t = threadIdx.x;
  float ze = 0.0f, ex = 0.0f;
  if (t < QLEN) {
    const unsigned* h = hist + (size_t)b * QLEN * NBINS_PAD + t * NBINS_PAD;
    float z1[5];
    #pragma unroll
    for (int j = 0; j < 5; ++j) z1[j] = b1[j];
    for (int k = 0; k < NBINS; ++k) {
      float hv = log1pf((float)h[k]);
      #pragma unroll
      for (int j = 0; j < 5; ++j) z1[j] += hv * W1[k * 5 + j];
    }
    float z2 = b2[0];
    #pragma unroll
    for (int j = 0; j < 5; ++j) z2 += tanhf(z1[j]) * W2[j];
    float z3 = tanhf(tanhf(z2) * W3[0] + b3[0]);
    ex = expf(gate[b * QLEN + t]) * qmk[b * QLEN + t];
    ze = z3 * ex;
  }
  #pragma unroll
  for (int m = 16; m >= 1; m >>= 1) {
    ze += __shfl_xor(ze, m, 32);
    ex += __shfl_xor(ex, m, 32);
  }
  if (t == 0) out[b] = ze / (ex + 1e-5f);
}

// ---------------------------------------------------------------------------
extern "C" void kernel_launch(void* const* d_in, const int* in_sizes, int n_in,
                              void* d_out, int out_size, void* d_ws, size_t ws_size,
                              hipStream_t stream) {
  const int*   query    = (const int*)  d_in[0];
  const int*   document = (const int*)  d_in[1];
  /* d_in[2] = q_idf : unused by the reference */
  const float* emb      = (const float*)d_in[3];
  const float* W1       = (const float*)d_in[4];
  const float* b1       = (const float*)d_in[5];
  const float* W2       = (const float*)d_in[6];
  const float* b2       = (const float*)d_in[7];
  const float* W3       = (const float*)d_in[8];
  const float* b3       = (const float*)d_in[9];
  const float* Wg       = (const float*)d_in[10];
  const float* bg       = (const float*)d_in[11];

  char* ws = (char*)d_ws;
  _Float16* qn = (_Float16*)ws;
  size_t off = (size_t)BATCH * QLEN * EPAD * sizeof(_Float16);   // 320 KB
  float* gate = (float*)(ws + off); off += (size_t)BATCH * QLEN * sizeof(float);
  float* qmk  = (float*)(ws + off); off += (size_t)BATCH * QLEN * sizeof(float);
  unsigned* hist = (unsigned*)(ws + off);                        // 64 KB

  drmm_prep<<<BATCH, 512, 0, stream>>>(query, emb, Wg, bg, qn, gate, qmk, hist);
  drmm_interact<<<dim3(DLEN / DTILE, BATCH), 128, 0, stream>>>(document, emb, qn, hist);
  drmm_head<<<BATCH, 32, 0, stream>>>(hist, gate, qmk, W1, b1, W2, b2, W3, b3,
                                      (float*)d_out);
}
